// EqModelComplex_14379550507644
// MI455X (gfx1250) — matrix operand order repaired and checked by reading, not compiled, not run on hardware
//
#include <hip/hip_runtime.h>
#include <hip/hip_bf16.h>

// ---------------- model dims (fixed by reference) ----------------
#define Bb    2
#define Ll    2048
#define Dd    512
#define Hh    8
#define HDd   64
#define HIDd  2048
#define BLn   (Bb*Ll)          // 4096 rows

typedef __bf16 bf16_t;
typedef __attribute__((ext_vector_type(16))) __bf16       v16bf;
typedef __attribute__((ext_vector_type(8)))  float        v8f;
typedef __attribute__((ext_vector_type(4)))  unsigned int u32x4;
typedef __attribute__((ext_vector_type(8)))  unsigned int u32x8;

// ---------------- WMMA helper ----------------
__device__ inline v8f wmma_bf16(v16bf a, v16bf b, v8f c) {
  // (neg_a, A, neg_b, B, c_mod, C, reuse_a, reuse_b)
  return __builtin_amdgcn_wmma_f32_16x16x32_bf16(false, a, false, b, (short)0, c,
                                                 false, false);
}

// A fragment: 16x32 bf16, row-major source with leading dim ld.
__device__ inline v16bf load_a_frag(const bf16_t* __restrict__ X, int m0, int k0, int ld) {
  const int lane = threadIdx.x & 31;
  const int hv = lane >> 4, l15 = lane & 15;
  const bf16_t* p = X + (size_t)(m0 + l15) * ld + k0 + hv * 8;
  union { v16bf v; u32x4 q[2]; } u;
  u.q[0] = *(const u32x4*)(p);
  u.q[1] = *(const u32x4*)(p + 16);
  return u.v;
}

// B fragment: 32x16 bf16 = (W[N,K])^T tile; lane holds two 8-elem chunks (k = 8h.., 16+8h..).
__device__ inline v16bf load_b_frag(const bf16_t* __restrict__ W, int n0, int k0, int ld) {
  const int lane = threadIdx.x & 31;
  const int hv = lane >> 4, l15 = lane & 15;
  const bf16_t* p = W + (size_t)(n0 + l15) * ld + k0 + hv * 8;
  union { v16bf v; u32x4 q[2]; } u;
  u.q[0] = *(const u32x4*)(p);
  u.q[1] = *(const u32x4*)(p + 16);
  return u.v;
}

// ---------------- Tensor Data Mover: 2D tile -> LDS (ISA ch.8 D#) -----------
// Group0: {count=1 | lds_addr | global_addr[56:0] | type=2}
// Group1: {data_size=2B | tensor_dim0/1 | tile_dim0/1 | tensor_dim0_stride}
__device__ inline void tdm_load_tile_2d(unsigned lds_byte_off, const void* gptr,
                                        unsigned tile_d0, unsigned tile_d1,
                                        unsigned tensor_d0, unsigned tensor_d1,
                                        unsigned stride0) {
  const unsigned long long ga = (unsigned long long)gptr;
  u32x4 g0;
  g0[0] = 1u;                                             // count=1 (valid)
  g0[1] = lds_byte_off;                                   // lds_addr
  g0[2] = (unsigned)ga;                                   // global_addr[31:0]
  g0[3] = (unsigned)((ga >> 32) & 0x01ffffffu) | (2u << 30);  // [56:32] | type=2
  u32x8 g1;
  g1[0] = 1u << 16;                                       // data_size=1 (2 bytes)
  g1[1] = tensor_d0 << 16;                                // tensor_dim0[15:0]
  g1[2] = (tensor_d0 >> 16) | (tensor_d1 << 16);          // dim0[31:16] | dim1[15:0]
  g1[3] = (tensor_d1 >> 16) | (tile_d0 << 16);            // dim1[31:16] | tile_dim0
  g1[4] = tile_d1 & 0xffffu;                              // tile_dim1 (tile_dim2=0)
  g1[5] = stride0;                                        // tensor_dim0_stride[31:0]
  g1[6] = 0u;
  g1[7] = 0u;
  asm volatile("tensor_load_to_lds %0, %1" :: "s"(g0), "s"(g1) : "memory");
}

// ---------------- f32 -> bf16 conversion ----------------
__global__ __launch_bounds__(256) void f32_to_bf16_kernel(const float* __restrict__ x,
                                                          bf16_t* __restrict__ y, int n) {
  int i = blockIdx.x * 256 + threadIdx.x;
  if (i < n) y[i] = (bf16_t)x[i];
}

// ---------------- complex LayerNorm (block per row, D=512) ----------------
__global__ __launch_bounds__(256) void cln_kernel(
    const float* __restrict__ xr, const float* __restrict__ xi,
    const float* __restrict__ gr, const float* __restrict__ gi,
    const float* __restrict__ br, const float* __restrict__ bi,
    bf16_t* __restrict__ nr, bf16_t* __restrict__ ni) {
  __shared__ float sm[3][8];
  const int tid = threadIdx.x, lane = tid & 31, wv = tid >> 5;
  const size_t base = (size_t)blockIdx.x * Dd;
  float a0 = xr[base + tid], a1 = xr[base + tid + 256];
  float b0 = xi[base + tid], b1 = xi[base + tid + 256];
  float sr = a0 + a1, si = b0 + b1;
#pragma unroll
  for (int m = 1; m < 32; m <<= 1) { sr += __shfl_xor(sr, m, 32); si += __shfl_xor(si, m, 32); }
  if (lane == 0) { sm[0][wv] = sr; sm[1][wv] = si; }
  __syncthreads();
  float tr = 0.f, ti = 0.f;
#pragma unroll
  for (int i = 0; i < 8; ++i) { tr += sm[0][i]; ti += sm[1][i]; }
  const float mr = tr * (1.0f / Dd), mi = ti * (1.0f / Dd);
  const float c0r = a0 - mr, c1r = a1 - mr, c0i = b0 - mi, c1i = b1 - mi;
  float sv = c0r * c0r + c1r * c1r + c0i * c0i + c1i * c1i;
#pragma unroll
  for (int m = 1; m < 32; m <<= 1) sv += __shfl_xor(sv, m, 32);
  if (lane == 0) sm[2][wv] = sv;
  __syncthreads();
  float tv = 0.f;
#pragma unroll
  for (int i = 0; i < 8; ++i) tv += sm[2][i];
  const float invs = 1.0f / sqrtf(tv * (1.0f / Dd) + 1e-6f);
#pragma unroll
  for (int e = 0; e < 2; ++e) {
    const int d = tid + e * 256;
    const float cr = (e ? c1r : c0r) * invs, ci = (e ? c1i : c0i) * invs;
    nr[base + d] = (bf16_t)(cr * gr[d] - ci * gi[d] + br[d]);
    ni[base + d] = (bf16_t)(cr * gi[d] + ci * gr[d] + bi[d]);
  }
}

// ---------------- complex GEMM:  Y = X (Wr + i Wi)^T  (+bias) (+residual) ----
__global__ __launch_bounds__(256) void cgemm_kernel(
    const bf16_t* __restrict__ Xr, const bf16_t* __restrict__ Xi,
    const bf16_t* __restrict__ Wr, const bf16_t* __restrict__ Wi,
    const float* __restrict__ br, const float* __restrict__ bi,
    const float* __restrict__ Rr, const float* __restrict__ Ri,
    float* __restrict__ Yr, float* __restrict__ Yi,
    int M, int N, int K) {
  const int wv = threadIdx.x >> 5, lane = threadIdx.x & 31;
  const int hv = lane >> 4, l15 = lane & 15;
  const int m0 = blockIdx.y * 64 + (wv >> 1) * 16;
  const int n0 = blockIdx.x * 64 + (wv & 1) * 32;
  const v8f zero = {0.f, 0.f, 0.f, 0.f, 0.f, 0.f, 0.f, 0.f};
  v8f accRR[2] = {zero, zero}, accII[2] = {zero, zero};
  v8f accRI[2] = {zero, zero}, accIR[2] = {zero, zero};
  for (int k = 0; k < K; k += 32) {
    if (k + 32 < K) {
      __builtin_prefetch(Xr + (size_t)(m0 + l15) * K + k + 32, 0, 1);
      __builtin_prefetch(Wr + (size_t)(n0 + l15) * K + k + 32, 0, 1);
    }
    v16bf ar = load_a_frag(Xr, m0, k, K);
    v16bf ai = load_a_frag(Xi, m0, k, K);
#pragma unroll
    for (int t = 0; t < 2; ++t) {
      v16bf bwr = load_b_frag(Wr, n0 + t * 16, k, K);
      v16bf bwi = load_b_frag(Wi, n0 + t * 16, k, K);
      accRR[t] = wmma_bf16(ar, bwr, accRR[t]);
      accII[t] = wmma_bf16(ai, bwi, accII[t]);
      accRI[t] = wmma_bf16(ar, bwi, accRI[t]);
      accIR[t] = wmma_bf16(ai, bwr, accIR[t]);
    }
  }
#pragma unroll
  for (int t = 0; t < 2; ++t) {
    const int col = n0 + t * 16 + l15;
    const float bvr = br ? br[col] : 0.f;
    const float bvi = bi ? bi[col] : 0.f;
#pragma unroll
    for (int r = 0; r < 8; ++r) {
      const int row = m0 + r + 8 * hv;
      const size_t o = (size_t)row * N + col;
      float yr = accRR[t][r] - accII[t][r] + bvr;
      float yi = accRI[t][r] + accIR[t][r] + bvi;
      if (Rr) { yr += Rr[o]; yi += Ri[o]; }
      Yr[o] = yr;
      Yi[o] = yi;
    }
  }
}

// ---------------- RoPE + head-pack: [B,L,D] f32 -> [B,H,L,HD] bf16 ----------
__global__ __launch_bounds__(256) void rope_pack_kernel(
    const float* __restrict__ qr, const float* __restrict__ qi,
    const float* __restrict__ kr, const float* __restrict__ ki,
    const float* __restrict__ cosb, const float* __restrict__ sinb,
    bf16_t* __restrict__ Qr, bf16_t* __restrict__ Qi,
    bf16_t* __restrict__ Kr, bf16_t* __restrict__ Ki) {
  const size_t i = (size_t)blockIdx.x * 256 + threadIdx.x;  // < B*L*D
  const int b = (int)(i / (Ll * Dd));
  const int rem = (int)(i % (Ll * Dd));
  const int l = rem / Dd, c = rem % Dd;
  const int h = c >> 6, d = c & 63, j = d & 31;
  const float cs = cosb[l * 32 + j], sn = sinb[l * 32 + j];
  const size_t rot = (d < 32) ? (i + 32) : (i - 32);
  const float sgn = (d < 32) ? -1.f : 1.f;
  const size_t o = ((size_t)(b * Hh + h) * Ll + l) * HDd + d;
  Qr[o] = (bf16_t)(qr[i] * cs + sgn * qr[rot] * sn);
  Qi[o] = (bf16_t)(qi[i] * cs + sgn * qi[rot] * sn);
  Kr[o] = (bf16_t)(kr[i] * cs + sgn * kr[rot] * sn);
  Ki[o] = (bf16_t)(ki[i] * cs + sgn * ki[rot] * sn);
}

// ---------------- V pack (transposed): [B,L,D] f32 -> [B,H,HD,L] bf16 -------
__global__ __launch_bounds__(256) void vpack_kernel(
    const float* __restrict__ vr, const float* __restrict__ vi,
    bf16_t* __restrict__ Vtr, bf16_t* __restrict__ Vti) {
  const size_t i = (size_t)blockIdx.x * 256 + threadIdx.x;
  const int b = (int)(i / (Ll * Dd));
  const int rem = (int)(i % (Ll * Dd));
  const int l = rem / Dd, c = rem % Dd;
  const int h = c >> 6, d = c & 63;
  const size_t o = ((size_t)(b * Hh + h) * HDd + d) * Ll + l;
  Vtr[o] = (bf16_t)vr[i];
  Vti[o] = (bf16_t)vi[i];
}

// ---------------- fused causal attention (flash-style, WMMA + TDM) ----------
// One wave = one 16-row q tile; block = 4 waves = 64 q rows.
// K/V tiles are DMA'd by the Tensor Data Mover into a double-buffered LDS
// stage (tracked with TENSORcnt); waves consume them via ds_load fragments.
//
// LDS pool layout (bytes; single shared object => allocated at LDS offset 0):
//   buf b in {0,1} at b*16384:  Kr[32][64] @ +0, Ki @ +4096, Vr[64][32] @ +8192,
//                               Vi @ +12288
//   P bounce tiles (per wave, 16x32 bf16) @ 32768
__global__ __launch_bounds__(128) void attn_kernel(
    const bf16_t* __restrict__ Qr, const bf16_t* __restrict__ Qi,
    const bf16_t* __restrict__ Kr, const bf16_t* __restrict__ Ki,
    const bf16_t* __restrict__ Vtr, const bf16_t* __restrict__ Vti,
    bf16_t* __restrict__ AOr, bf16_t* __restrict__ AOi) {
  __shared__ bf16_t smem[18432];  // 36 KB: 2x16KB KV stage + 4KB P bounce
  const int wv = threadIdx.x >> 5, lane = threadIdx.x & 31;
  const int hv = lane >> 4, l15 = lane & 15;
  const int bh = blockIdx.x / (Ll / 64), qblk = blockIdx.x % (Ll / 64);
  const int b = bh >> 3, h = bh & 7;
  const bf16_t* Qrp = Qr + (size_t)bh * Ll * HDd;
  const bf16_t* Qip = Qi + (size_t)bh * Ll * HDd;
  const bf16_t* Krp = Kr + (size_t)bh * Ll * HDd;
  const bf16_t* Kip = Ki + (size_t)bh * Ll * HDd;
  const bf16_t* Vrp = Vtr + (size_t)bh * HDd * Ll;
  const bf16_t* Vip = Vti + (size_t)bh * HDd * Ll;
  const int q0 = qblk * 64 + wv * 16;
  const float scale = 0.125f;  // HD^-0.5

  v16bf aqr[2], aqi[2];
#pragma unroll
  for (int s = 0; s < 2; ++s) {
    aqr[s] = load_a_frag(Qrp, q0, s * 32, HDd);
    aqi[s] = load_a_frag(Qip, q0, s * 32, HDd);
  }
  const v8f zero = {0.f, 0.f, 0.f, 0.f, 0.f, 0.f, 0.f, 0.f};
  v8f o[2][4];
  float mrow[8], lrow[8];
#pragma unroll
  for (int r = 0; r < 8; ++r) { mrow[r] = -1e30f; lrow[r] = 0.f; }
#pragma unroll
  for (int pt = 0; pt < 2; ++pt)
#pragma unroll
    for (int dt = 0; dt < 4; ++dt) o[pt][dt] = zero;

  const int kend = qblk * 64 + 64;  // block-uniform trip count
  // ---- TDM preload of tile kb=0 into buffer 0 ----
  if (wv == 0) {
    tdm_load_tile_2d(0u, Krp, 64, 32, 64, 32, 64);
    tdm_load_tile_2d(4096u, Kip, 64, 32, 64, 32, 64);
    tdm_load_tile_2d(8192u, Vrp, 32, 64, Ll, 64, Ll);
    tdm_load_tile_2d(12288u, Vip, 32, 64, Ll, 64, Ll);
  }
  __builtin_amdgcn_s_wait_tensorcnt(0);
  __syncthreads();

  unsigned cur = 0;
  for (int kb = 0; kb < kend; kb += 32) {
    // ---- async: DMA next KV tile into the other buffer ----
    if (wv == 0 && kb + 32 < kend) {
      const unsigned nb = (cur ^ 1u) * 16384u;
      tdm_load_tile_2d(nb + 0u, Krp + (size_t)(kb + 32) * HDd, 64, 32, 64, 32, 64);
      tdm_load_tile_2d(nb + 4096u, Kip + (size_t)(kb + 32) * HDd, 64, 32, 64, 32, 64);
      tdm_load_tile_2d(nb + 8192u, Vrp + (kb + 32), 32, 64, Ll, 64, Ll);
      tdm_load_tile_2d(nb + 12288u, Vip + (kb + 32), 32, 64, Ll, 64, Ll);
    }
    if (kb < q0 + 16) {  // wave-uniform causal skip
      const bf16_t* kR = smem + (size_t)cur * 8192;  // element offsets
      const bf16_t* kI = kR + 2048;
      const bf16_t* vR = kR + 4096;
      const bf16_t* vI = kR + 6144;
      bf16_t* pb = smem + 16384 + wv * 512;  // this wave's P tile [16][32]
      // ---- scores: S[16q x 32k] = Qr.Kr^T + Qi.Ki^T (from LDS stage) ----
      v8f sAcc[2] = {zero, zero};
#pragma unroll
      for (int sd = 0; sd < 2; ++sd) {
#pragma unroll
        for (int t = 0; t < 2; ++t) {
          v16bf bkr = load_b_frag(kR, t * 16, sd * 32, HDd);
          sAcc[t] = wmma_bf16(aqr[sd], bkr, sAcc[t]);
          v16bf bki = load_b_frag(kI, t * 16, sd * 32, HDd);
          sAcc[t] = wmma_bf16(aqi[sd], bki, sAcc[t]);
        }
      }
      // ---- online softmax ----
      float mnew[8], ps[8];
#pragma unroll
      for (int r = 0; r < 8; ++r) {
        const int qrow = q0 + r + 8 * hv;
        float mx = -1e30f;
#pragma unroll
        for (int t = 0; t < 2; ++t) {
          float svv = sAcc[t][r] * scale;
          sAcc[t][r] = svv;
          if (kb + t * 16 + l15 <= qrow) mx = fmaxf(mx, svv);
        }
#pragma unroll
        for (int m = 1; m < 16; m <<= 1) mx = fmaxf(mx, __shfl_xor(mx, m, 32));
        mnew[r] = fmaxf(mrow[r], mx);
        ps[r] = 0.f;
      }
#pragma unroll
      for (int t = 0; t < 2; ++t)
#pragma unroll
        for (int r = 0; r < 8; ++r) {
          const int qrow = q0 + r + 8 * hv;
          const float p =
              (kb + t * 16 + l15 <= qrow) ? __expf(sAcc[t][r] - mnew[r]) : 0.f;
          sAcc[t][r] = p;
          ps[r] += p;
        }
#pragma unroll
      for (int r = 0; r < 8; ++r) {
#pragma unroll
        for (int m = 1; m < 16; m <<= 1) ps[r] += __shfl_xor(ps[r], m, 32);
        const float alpha = __expf(mrow[r] - mnew[r]);
        lrow[r] = lrow[r] * alpha + ps[r];
        mrow[r] = mnew[r];
#pragma unroll
        for (int pt = 0; pt < 2; ++pt)
#pragma unroll
          for (int dt = 0; dt < 4; ++dt) o[pt][dt][r] *= alpha;
      }
      // ---- P: C-layout -> bf16 A-layout via per-wave LDS tile ----
#pragma unroll
      for (int t = 0; t < 2; ++t)
#pragma unroll
        for (int r = 0; r < 8; ++r)
          pb[(r + 8 * hv) * 32 + t * 16 + l15] = (bf16_t)sAcc[t][r];
      __asm volatile("" ::: "memory");  // wave-internal; DS pipe is in-order
      v16bf ap;
      {
        union { v16bf v; u32x4 q[2]; } u;
        const bf16_t* pp = pb + l15 * 32 + hv * 8;
        u.q[0] = *(const u32x4*)(pp);
        u.q[1] = *(const u32x4*)(pp + 16);
        ap = u.v;
      }
      // ---- O += P.V (V tile staged transposed: [64][32], ld = 32) ----
#pragma unroll
      for (int dt = 0; dt < 4; ++dt) {
        v16bf bvr = load_b_frag(vR, dt * 16, 0, 32);
        o[0][dt] = wmma_bf16(ap, bvr, o[0][dt]);
        v16bf bvi = load_b_frag(vI, dt * 16, 0, 32);
        o[1][dt] = wmma_bf16(ap, bvi, o[1][dt]);
      }
    }
    // ---- pipeline: next DMA must be complete before buffers swap ----
    __builtin_amdgcn_s_wait_tensorcnt(0);
    __syncthreads();
    cur ^= 1u;
  }
  // ---- normalize + write merged-head [B,L,D] bf16 ----
#pragma unroll
  for (int r = 0; r < 8; ++r) {
    const int qrow = q0 + r + 8 * hv;
    const float inv = 1.0f / lrow[r];
#pragma unroll
    for (int dt = 0; dt < 4; ++dt) {
      const size_t off = (size_t)(b * Ll + qrow) * Dd + h * HDd + dt * 16 + l15;
      AOr[off] = (bf16_t)(o[0][dt][r] * inv);
      AOi[off] = (bf16_t)(o[1][dt][r] * inv);
    }
  }
}

// ---------------- ModReLU: relu(|z|+b) * z/|z| ------------------------------
__global__ __launch_bounds__(256) void modrelu_kernel(
    const float* __restrict__ hr, const float* __restrict__ hi,
    const float* __restrict__ modb, bf16_t* __restrict__ orr,
    bf16_t* __restrict__ oii) {
  const size_t i = (size_t)blockIdx.x * 256 + threadIdx.x;  // < 4096*2048
  const int col = (int)(i & (HIDd - 1));
  const float a = hr[i], bv = hi[i];
  const float mag = sqrtf(a * a + bv * bv);
  const float act = fmaxf(mag + modb[col], 0.f);
  float rr, ii;
  if (mag > 0.f) {
    const float s = act / mag;
    rr = a * s;
    ii = bv * s;
  } else {
    rr = act;  // angle(0,0)=0 -> cos=1, sin=0
    ii = 0.f;
  }
  orr[i] = (bf16_t)rr;
  oii[i] = (bf16_t)ii;
}

// ---------------- launcher ----------------
extern "C" void kernel_launch(void* const* d_in, const int* in_sizes, int n_in,
                              void* d_out, int out_size, void* d_ws, size_t ws_size,
                              hipStream_t stream) {
  (void)in_sizes; (void)n_in; (void)out_size; (void)ws_size;
  const float* x_real = (const float*)d_in[0];
  const float* x_imag = (const float*)d_in[1];
  const float* bo_r = (const float*)d_in[10];
  const float* bo_i = (const float*)d_in[11];
  const float* g1_r = (const float*)d_in[12];
  const float* g1_i = (const float*)d_in[13];
  const float* be1_r = (const float*)d_in[14];
  const float* be1_i = (const float*)d_in[15];
  const float* g2_r = (const float*)d_in[16];
  const float* g2_i = (const float*)d_in[17];
  const float* be2_r = (const float*)d_in[18];
  const float* be2_i = (const float*)d_in[19];
  const float* b1_r = (const float*)d_in[22];
  const float* b1_i = (const float*)d_in[23];
  const float* mod_b = (const float*)d_in[24];
  const float* b2_r = (const float*)d_in[27];
  const float* b2_i = (const float*)d_in[28];
  const float* cosb = (const float*)d_in[29];
  const float* sinb = (const float*)d_in[30];

  char* ws = (char*)d_ws;
  const size_t MBy = 1ull << 20, KBy = 1024;
  // activation arena
  bf16_t* n1r = (bf16_t*)(ws + 0 * MBy);
  bf16_t* n1i = (bf16_t*)(ws + 4 * MBy);
  float* qr_f = (float*)(ws + 8 * MBy);
  float* qi_f = (float*)(ws + 16 * MBy);
  float* kr_f = (float*)(ws + 24 * MBy);
  float* ki_f = (float*)(ws + 32 * MBy);
  float* vr_f = (float*)(ws + 40 * MBy);
  float* vi_f = (float*)(ws + 48 * MBy);
  bf16_t* Qrb = (bf16_t*)(ws + 56 * MBy);
  bf16_t* Qib = (bf16_t*)(ws + 60 * MBy);
  bf16_t* Krb = (bf16_t*)(ws + 64 * MBy);
  bf16_t* Kib = (bf16_t*)(ws + 68 * MBy);
  bf16_t* Vtr = (bf16_t*)(ws + 72 * MBy);
  bf16_t* Vti = (bf16_t*)(ws + 76 * MBy);
  bf16_t* AOr = (bf16_t*)(ws + 80 * MBy);
  bf16_t* AOi = (bf16_t*)(ws + 84 * MBy);
  float* x1r = (float*)(ws + 88 * MBy);
  float* x1i = (float*)(ws + 96 * MBy);
  float* hr_f = (float*)(ws + 8 * MBy);   // reuse: qkv f32 dead by then
  float* hi_f = (float*)(ws + 40 * MBy);  // reuse
  bf16_t* hbr = (bf16_t*)(ws + 104 * MBy);
  bf16_t* hbi = (bf16_t*)(ws + 120 * MBy);
  // bf16 weights
  const size_t WB = 136 * MBy;
  bf16_t* wqr = (bf16_t*)(ws + WB + 0 * 512 * KBy);
  bf16_t* wqi = (bf16_t*)(ws + WB + 1 * 512 * KBy);
  bf16_t* wkr = (bf16_t*)(ws + WB + 2 * 512 * KBy);
  bf16_t* wki = (bf16_t*)(ws + WB + 3 * 512 * KBy);
  bf16_t* wvr = (bf16_t*)(ws + WB + 4 * 512 * KBy);
  bf16_t* wvi = (bf16_t*)(ws + WB + 5 * 512 * KBy);
  bf16_t* wor = (bf16_t*)(ws + WB + 6 * 512 * KBy);
  bf16_t* woi = (bf16_t*)(ws + WB + 7 * 512 * KBy);
  bf16_t* w1r = (bf16_t*)(ws + WB + 4 * MBy);
  bf16_t* w1i = (bf16_t*)(ws + WB + 6 * MBy);
  bf16_t* w2r = (bf16_t*)(ws + WB + 8 * MBy);
  bf16_t* w2i = (bf16_t*)(ws + WB + 10 * MBy);

  // weight f32 -> bf16
  struct WC { int idx; bf16_t* dst; int n; };
  const WC wc[12] = {
      {2, wqr, Dd * Dd},  {3, wqi, Dd * Dd},  {4, wkr, Dd * Dd},
      {5, wki, Dd * Dd},  {6, wvr, Dd * Dd},  {7, wvi, Dd * Dd},
      {8, wor, Dd * Dd},  {9, woi, Dd * Dd},  {20, w1r, HIDd * Dd},
      {21, w1i, HIDd * Dd}, {25, w2r, Dd * HIDd}, {26, w2i, Dd * HIDd}};
  for (int i = 0; i < 12; ++i)
    f32_to_bf16_kernel<<<(wc[i].n + 255) / 256, 256, 0, stream>>>(
        (const float*)d_in[wc[i].idx], wc[i].dst, wc[i].n);

  // LN1
  cln_kernel<<<BLn, 256, 0, stream>>>(x_real, x_imag, g1_r, g1_i, be1_r, be1_i,
                                      n1r, n1i);
  // QKV projections
  const dim3 gP(Dd / 64, BLn / 64);
  cgemm_kernel<<<gP, 256, 0, stream>>>(n1r, n1i, wqr, wqi, nullptr, nullptr,
                                       nullptr, nullptr, qr_f, qi_f, BLn, Dd, Dd);
  cgemm_kernel<<<gP, 256, 0, stream>>>(n1r, n1i, wkr, wki, nullptr, nullptr,
                                       nullptr, nullptr, kr_f, ki_f, BLn, Dd, Dd);
  cgemm_kernel<<<gP, 256, 0, stream>>>(n1r, n1i, wvr, wvi, nullptr, nullptr,
                                       nullptr, nullptr, vr_f, vi_f, BLn, Dd, Dd);
  // RoPE + packing
  rope_pack_kernel<<<(Bb * Ll * Dd) / 256, 256, 0, stream>>>(
      qr_f, qi_f, kr_f, ki_f, cosb, sinb, Qrb, Qib, Krb, Kib);
  vpack_kernel<<<(Bb * Ll * Dd) / 256, 256, 0, stream>>>(vr_f, vi_f, Vtr, Vti);
  // fused causal attention (TDM-staged KV)
  attn_kernel<<<Bb * Hh * (Ll / 64), 128, 0, stream>>>(Qrb, Qib, Krb, Kib, Vtr,
                                                       Vti, AOr, AOi);
  // out projection + bias + residual(x)
  cgemm_kernel<<<gP, 256, 0, stream>>>(AOr, AOi, wor, woi, bo_r, bo_i, x_real,
                                       x_imag, x1r, x1i, BLn, Dd, Dd);
  // LN2 (reuse n1 buffers)
  cln_kernel<<<BLn, 256, 0, stream>>>(x1r, x1i, g2_r, g2_i, be2_r, be2_i, n1r,
                                      n1i);
  // FFN up + bias
  cgemm_kernel<<<dim3(HIDd / 64, BLn / 64), 256, 0, stream>>>(
      n1r, n1i, w1r, w1i, b1_r, b1_i, nullptr, nullptr, hr_f, hi_f, BLn, HIDd, Dd);
  // ModReLU
  modrelu_kernel<<<(BLn * HIDd) / 256, 256, 0, stream>>>(hr_f, hi_f, mod_b, hbr,
                                                         hbi);
  // FFN down + bias + residual(x1) -> d_out (out_r then out_i)
  float* out_r = (float*)d_out;
  float* out_i = out_r + (size_t)BLn * Dd;
  cgemm_kernel<<<gP, 256, 0, stream>>>(hbr, hbi, w2r, w2i, b2_r, b2_i, x1r, x1i,
                                       out_r, out_i, BLn, Dd, HIDd);
}
